// AAConv2d_27797028340277
// MI455X (gfx1250) — matrix operand order repaired and checked by reading, not compiled, not run on hardware
//
#include <hip/hip_runtime.h>
#include <hip/hip_bf16.h>

// AAConv2d for MI455X (gfx1250): all matmuls via v_wmma_f32_16x16x32_f16.
// B=8, CIN=128, H=W=32, COUT=128, DK=DV=64, NH=8, DKH=DVH=8, L=1024.
// All operands pre-converted to f16 once; hot loops do pure b128 f16 loads.

typedef __attribute__((ext_vector_type(16))) _Float16 v16h;
typedef __attribute__((ext_vector_type(8)))  float    v8f;

#define NB    8
#define CIN   128
#define HW    1024     // 32*32
#define LQ    1024     // sequence length per head
#define NH    8
#define DKH   8
#define SQRT8 2.82842712474619f

union PackH8 { _Float16 h[8]; uint4 u; };

// ---------------------------------------------------------------------------
// Pre-pass: fp32 -> f16 (vectorized 8 elements/thread). n must be mult. of 8.
// ---------------------------------------------------------------------------
__global__ __launch_bounds__(256) void cvt_f32_to_f16_kernel(
    const float* __restrict__ src, _Float16* __restrict__ dst, int n8)
{
    int i = blockIdx.x * 256 + threadIdx.x;
    if (i < n8) {
        const float4 a = ((const float4*)src)[2 * i];
        const float4 b = ((const float4*)src)[2 * i + 1];
        PackH8 pk;
        pk.h[0] = (_Float16)a.x; pk.h[1] = (_Float16)a.y;
        pk.h[2] = (_Float16)a.z; pk.h[3] = (_Float16)a.w;
        pk.h[4] = (_Float16)b.x; pk.h[5] = (_Float16)b.y;
        pk.h[6] = (_Float16)b.z; pk.h[7] = (_Float16)b.w;
        ((uint4*)dst)[i] = pk.u;
    }
}

// ---------------------------------------------------------------------------
// Kernel 1: qkv 1x1 conv (GEMM M=8192, N=192, K=128), f16 in -> f16 q/k/v.
// Grid.x = 512 M-tiles * 3 groups; group g in {0,1,2} = {q,k,v} (64 ch each).
// Output layout: [b][head][dim][l] (dim-major) -> contiguous b128 stores.
// ---------------------------------------------------------------------------
__global__ __launch_bounds__(32) void qkv_gemm_kernel(
    const _Float16* __restrict__ x16, const _Float16* __restrict__ w16,
    const float* __restrict__ bias,
    _Float16* __restrict__ q16, _Float16* __restrict__ k16,
    _Float16* __restrict__ v16)
{
    __shared__ _Float16 sA[16 * 32];
    const int lane  = threadIdx.x;
    const int mt    = blockIdx.x & 511;    // M tile
    const int g     = blockIdx.x >> 9;     // 0=q, 1=k, 2=v
    const int Mbase = mt * 16;
    const int Nbase = g * 64;

    _Float16* const dst  = (g == 0) ? q16 : (g == 1) ? k16 : v16;
    const float     scal = (g == 0) ? SQRT8 : 1.0f;

    v8f cacc[4];
    #pragma unroll
    for (int t = 0; t < 4; ++t) cacc[t] = (v8f){};

    for (int kb = 0; kb < CIN; kb += 32) {
        // stage A tile (16 rows x 32 k) into LDS (f16 passthrough, no cvt)
        #pragma unroll
        for (int e = 0; e < 16; ++e) {
            int idx = lane + 32 * e;              // 0..511
            int m = idx & 15, kk = idx >> 4;      // kk 0..31
            int M = Mbase + m;
            int b = M >> 10, hw = M & 1023;
            sA[m * 32 + kk] = x16[(b * CIN + kb + kk) * HW + hw];
        }
        __syncthreads();
        // A fragment (ISA 16-bit A 16x32 layout)
        v16h a;
        {
            int m  = lane & 15;
            int kx = (lane < 16) ? 0 : 8;
            #pragma unroll
            for (int e = 0; e < 8; ++e) a[e]     = sA[m * 32 + kx + e];
            #pragma unroll
            for (int e = 0; e < 8; ++e) a[8 + e] = sA[m * 32 + 16 + kx + e];
        }
        // 4 B fragments (w16 row-major [192][128]) + 4 WMMA, A reused
        const int k2 = kb + ((lane < 16) ? 0 : 8);
        #pragma unroll
        for (int t = 0; t < 4; ++t) {
            int n = Nbase + t * 16 + (lane & 15);
            const _Float16* wr = w16 + (size_t)n * CIN + k2;
            __builtin_prefetch(wr + 32, 0, 3);
            v16h bf;                      // 16 halfs = 2 aligned uint4 loads
            {
                uint4 u0 = *(const uint4*)(wr);
                uint4 u1 = *(const uint4*)(wr + 16);
                const _Float16* h0 = (const _Float16*)&u0;
                const _Float16* h1 = (const _Float16*)&u1;
                #pragma unroll
                for (int e = 0; e < 8; ++e) { bf[e] = h0[e]; bf[8 + e] = h1[e]; }
            }
            cacc[t] = __builtin_amdgcn_wmma_f32_16x16x32_f16(
                false, a, false, bf, (short)0, cacc[t], false, false);
        }
        __syncthreads();
    }
    // epilogue: 8 contiguous hw per lane -> one b128 store per tile
    const int rbase = (lane < 16) ? 0 : 8;
    const int M0    = Mbase + rbase;
    const int b     = M0 >> 10, hw0 = M0 & 1023;   // hw0..hw0+7 same image
    #pragma unroll
    for (int t = 0; t < 4; ++t) {
        int   nl   = t * 16 + (lane & 15);         // 0..63 within group
        int   head = nl >> 3, d = nl & 7;
        float bv   = bias[Nbase + nl];
        PackH8 pk;
        #pragma unroll
        for (int v = 0; v < 8; ++v)
            pk.h[v] = (_Float16)((cacc[t][v] + bv) * scal);
        *(uint4*)(dst + (((size_t)(b * NH + head) * DKH + d) << 10) + hw0) = pk.u;
    }
}

// ---------------------------------------------------------------------------
// Kernel 2: fused attention per (b,head). Two-pass softmax, logits stay in
// the WGP. Whole-head K and V resident in LDS (CDNA5: 320KB/WGP).
// logits[n,m] = q[n].k[m] + q[n].rel_h[h2-h1+31] + q[n].rel_w[w2-w1+31]
// Row sums via ones-column appended to the V B-fragment (col 8 of acc).
// q/k/v arrive in [b][head][dim][l] layout; output written as f16.
// ---------------------------------------------------------------------------
__global__ __launch_bounds__(256) void attn_kernel(
    const _Float16* __restrict__ q16, const _Float16* __restrict__ k16,
    const _Float16* __restrict__ v16,
    const float* __restrict__ rel_h, const float* __restrict__ rel_w,
    _Float16* __restrict__ attn16)
{
    __shared__ _Float16 sK [LQ * DKH];     // [key][dim]   16 KB
    __shared__ _Float16 sVT[DKH * LQ];     // [dim][key]   16 KB
    __shared__ _Float16 sRH[8 * 16 * 32];  // [wave][q][h2] 8 KB
    __shared__ _Float16 sRW[8 * 16 * 32];  // [wave][q][w2] 8 KB
    __shared__ _Float16 sP [8 * 16 * 32];  // per-wave 16x32 P tile 8 KB

    const int tid  = threadIdx.x;
    const int lane = tid & 31, wave = tid >> 5;
    const int bh   = blockIdx.x >> 3;      // (b,head) 0..63
    const int qblk = blockIdx.x & 7;
    const int b    = bh >> 3, head = bh & 7;

    const _Float16* qg = q16 + (size_t)bh * DKH * LQ;  // [dim][l]
    const _Float16* kg = k16 + (size_t)bh * DKH * LQ;
    const _Float16* vg = v16 + (size_t)bh * DKH * LQ;

    // K: transpose [dim][key] -> [key][dim]; coalesced b16 loads, b128 LDS st
    for (int key = tid; key < LQ; key += 256) {
        PackH8 pk;
        #pragma unroll
        for (int d = 0; d < DKH; ++d) pk.h[d] = kg[(d << 10) + key];
        *(uint4*)(sK + key * DKH) = pk.u;
    }
    // V^T: already dim-major -> straight vector copy
    {
        const uint4* src = (const uint4*)vg;
        uint4*       dstl = (uint4*)sVT;
        for (int i = tid; i < (LQ * DKH) / 8; i += 256) dstl[i] = src[i];
    }

    // relative-logit tables for this wave's 16 queries
    const int qbase = qblk * 128 + wave * 16;
    for (int i = 0; i < 32; ++i) {
        int id = lane + 32 * i;            // 0..1023
        int qi = id >> 6;                  // query 0..15
        int r  = id & 63;                  // 0..31 -> rh, 32..63 -> rw
        int qglob = qbase + qi;
        int h1 = qglob >> 5, w1 = qglob & 31;
        float acc = 0.f;
        if (r < 32) {
            const float* kr = rel_h + (size_t)(r - h1 + 31) * DKH;
            #pragma unroll
            for (int d = 0; d < DKH; ++d)
                acc += (float)qg[(d << 10) + qglob] * kr[d];
            sRH[(wave * 16 + qi) * 32 + r] = (_Float16)acc;
        } else {
            int w2 = r - 32;
            const float* kr = rel_w + (size_t)(w2 - w1 + 31) * DKH;
            #pragma unroll
            for (int d = 0; d < DKH; ++d)
                acc += (float)qg[(d << 10) + qglob] * kr[d];
            sRW[(wave * 16 + qi) * 32 + w2] = (_Float16)acc;
        }
    }
    __syncthreads();

    // Q A-fragment: dkh=8 padded to K=32 with zeros
    v16h aq = {};
    if (lane < 16) {
        #pragma unroll
        for (int e = 0; e < 8; ++e) aq[e] = qg[(e << 10) + qbase + lane];
    }
    const _Float16* rhW = sRH + wave * 16 * 32;
    const _Float16* rwW = sRW + wave * 16 * 32;
    const int col   = lane & 15;
    const int rbase = (lane < 16) ? 0 : 8;

    // ---- pass 1: row maxima over all 1024 keys ----
    float rowmax[8];
    #pragma unroll
    for (int v = 0; v < 8; ++v) rowmax[v] = -1e30f;
    for (int j = 0; j < 64; ++j) {
        v16h bk = {};
        if (lane < 16) {
            const uint4 kv = *(const uint4*)(sK + (size_t)(j * 16 + lane) * DKH);
            const _Float16* kh = (const _Float16*)&kv;
            #pragma unroll
            for (int e = 0; e < 8; ++e) bk[e] = kh[e];
        }
        v8f c = {};
        c = __builtin_amdgcn_wmma_f32_16x16x32_f16(false, aq, false, bk,
                                                   (short)0, c, false, false);
        int key = j * 16 + col;
        int h2 = key >> 5, w2 = key & 31;
        #pragma unroll
        for (int v = 0; v < 8; ++v) {
            int row = rbase + v;
            float l = c[v] + (float)rhW[row * 32 + h2] + (float)rwW[row * 32 + w2];
            rowmax[v] = fmaxf(rowmax[v], l);
        }
    }
    // cross-lane max within each 16-lane half (rows identical across a half)
    #pragma unroll
    for (int v = 0; v < 8; ++v)
        #pragma unroll
        for (int m = 1; m < 16; m <<= 1)
            rowmax[v] = fmaxf(rowmax[v], __shfl_xor(rowmax[v], m, 32));

    // ---- pass 2: P = exp(l - max), acc = P @ [V | 1] ----
    v8f acc = {};
    for (int jj = 0; jj < 32; ++jj) {
        #pragma unroll
        for (int t = 0; t < 2; ++t) {
            int j = jj * 2 + t;
            v16h bk = {};
            if (lane < 16) {
                const uint4 kv = *(const uint4*)(sK + (size_t)(j * 16 + lane) * DKH);
                const _Float16* kh = (const _Float16*)&kv;
                #pragma unroll
                for (int e = 0; e < 8; ++e) bk[e] = kh[e];
            }
            v8f c = {};
            c = __builtin_amdgcn_wmma_f32_16x16x32_f16(false, aq, false, bk,
                                                       (short)0, c, false, false);
            int key = j * 16 + col;
            int h2 = key >> 5, w2 = key & 31;
            #pragma unroll
            for (int v = 0; v < 8; ++v) {
                int row = rbase + v;
                float l = c[v] + (float)rhW[row * 32 + h2] + (float)rwW[row * 32 + w2];
                sP[(wave * 16 + row) * 32 + t * 16 + col] =
                    (_Float16)__expf(l - rowmax[v]);
            }
        }
        __syncthreads();   // cross-lane LDS transpose hazard (uniform loop)
        v16h ap;
        {
            int m  = lane & 15;
            int kx = (lane < 16) ? 0 : 16;
            const _Float16* pr = sP + (wave * 16 + m) * 32;
            #pragma unroll
            for (int e = 0; e < 8; ++e) ap[e]     = pr[kx + e];
            #pragma unroll
            for (int e = 0; e < 8; ++e) ap[8 + e] = pr[kx + 8 + e];
        }
        v16h bv;
        {
            int k2 = jj * 32 + ((lane < 16) ? 0 : 16);
            if (col < 8) {
                const _Float16* vr = sVT + col * LQ + k2;
                #pragma unroll
                for (int e = 0; e < 16; ++e) bv[e] = vr[e];
            } else if (col == 8) {
                #pragma unroll
                for (int e = 0; e < 16; ++e) bv[e] = (_Float16)1.0f;
            } else {
                #pragma unroll
                for (int e = 0; e < 16; ++e) bv[e] = (_Float16)0.0f;
            }
        }
        acc = __builtin_amdgcn_wmma_f32_16x16x32_f16(false, ap, false, bv,
                                                     (short)0, acc, false, false);
        __syncthreads();
    }

    // normalize by row sum (held in column 8 of the accumulator tile)
    #pragma unroll
    for (int v = 0; v < 8; ++v) {
        float s = __shfl(acc[v], (lane & 16) + 8, 32);
        float o = acc[v] / s;
        if (col < 8) {
            int row   = rbase + v;
            int query = qbase + row;
            attn16[((size_t)(b * 64 + head * 8 + col)) * LQ + query] = (_Float16)o;
        }
    }
}

// ---------------------------------------------------------------------------
// Kernel 3: 3x3 conv (pad 1) as implicit GEMM: M=8192, N=64, K=cin*9.
// f16 input & weights; one wave computes a 16x64 output strip: A staged once
// per K-step, 4 B fragments + 4 WMMA (A reuse). Grid.x = 512 M-tiles.
// ---------------------------------------------------------------------------
__global__ __launch_bounds__(32) void conv3x3_wmma_kernel(
    const _Float16* __restrict__ x16, int cin,
    const _Float16* __restrict__ w16, const float* __restrict__ bias,
    float* __restrict__ out, int out_ch_base)
{
    __shared__ _Float16 sA[16 * 32];
    const int lane  = threadIdx.x;
    const int K     = cin * 9;
    const int Mbase = (int)blockIdx.x * 16;

    v8f cacc[4];
    #pragma unroll
    for (int t = 0; t < 4; ++t) cacc[t] = (v8f){};

    for (int kb = 0; kb < K; kb += 32) {
        #pragma unroll
        for (int e = 0; e < 16; ++e) {
            int idx = lane + 32 * e;
            int m = idx & 15, kk = idx >> 4;
            int M = Mbase + m;
            int b = M >> 10, hw = M & 1023;
            int h = hw >> 5, ww = hw & 31;
            int k  = kb + kk;
            int ci = k / 9, r = k - ci * 9;
            int kh = r / 3, kw = r - kh * 3;
            int hh = h + kh - 1, w2 = ww + kw - 1;
            _Float16 val = (_Float16)0.0f;
            if ((unsigned)hh < 32u && (unsigned)w2 < 32u)
                val = x16[((size_t)(b * cin + ci)) * HW + hh * 32 + w2];
            sA[m * 32 + kk] = val;
        }
        __syncthreads();
        v16h a;
        {
            int m  = lane & 15;
            int kx = (lane < 16) ? 0 : 8;
            #pragma unroll
            for (int e = 0; e < 8; ++e) a[e]     = sA[m * 32 + kx + e];
            #pragma unroll
            for (int e = 0; e < 8; ++e) a[8 + e] = sA[m * 32 + 16 + kx + e];
        }
        const int k2 = kb + ((lane < 16) ? 0 : 8);
        #pragma unroll
        for (int t = 0; t < 4; ++t) {
            int n = t * 16 + (lane & 15);
            const _Float16* wr = w16 + (size_t)n * K + k2;
            __builtin_prefetch(wr + 32, 0, 3);
            v16h bf;                      // 16 halfs = 2 aligned uint4 loads
            {
                uint4 u0 = *(const uint4*)(wr);
                uint4 u1 = *(const uint4*)(wr + 16);
                const _Float16* h0 = (const _Float16*)&u0;
                const _Float16* h1 = (const _Float16*)&u1;
                #pragma unroll
                for (int e = 0; e < 8; ++e) { bf[e] = h0[e]; bf[8 + e] = h1[e]; }
            }
            cacc[t] = __builtin_amdgcn_wmma_f32_16x16x32_f16(
                false, a, false, bf, (short)0, cacc[t], false, false);
        }
        __syncthreads();
    }
    // epilogue: rows are 8 contiguous hw -> two b128 (float4) stores per tile
    const int rbase = (lane < 16) ? 0 : 8;
    const int M0    = Mbase + rbase;
    const int b     = M0 >> 10, hw0 = M0 & 1023;
    #pragma unroll
    for (int t = 0; t < 4; ++t) {
        int   n  = t * 16 + (lane & 15);
        float bv = bias[n];
        float* op = out + ((size_t)(b * 128 + out_ch_base + n)) * HW + hw0;
        float4 lo, hi;
        lo.x = cacc[t][0] + bv; lo.y = cacc[t][1] + bv;
        lo.z = cacc[t][2] + bv; lo.w = cacc[t][3] + bv;
        hi.x = cacc[t][4] + bv; hi.y = cacc[t][5] + bv;
        hi.z = cacc[t][6] + bv; hi.w = cacc[t][7] + bv;
        *(float4*)(op)     = lo;
        *(float4*)(op + 4) = hi;
    }
}

// ---------------------------------------------------------------------------
extern "C" void kernel_launch(void* const* d_in, const int* in_sizes, int n_in,
                              void* d_out, int out_size, void* d_ws, size_t ws_size,
                              hipStream_t stream) {
    (void)in_sizes; (void)n_in; (void)out_size; (void)ws_size;
    const float* x         = (const float*)d_in[0];
    const float* w_general = (const float*)d_in[1];
    const float* b_general = (const float*)d_in[2];
    const float* w_qkv     = (const float*)d_in[3];
    const float* b_qkv     = (const float*)d_in[4];
    const float* w_out     = (const float*)d_in[5];
    const float* b_out     = (const float*)d_in[6];
    const float* rel_h     = (const float*)d_in[7];
    const float* rel_w     = (const float*)d_in[8];
    float* out = (float*)d_out;

    char* ws = (char*)d_ws;
    _Float16* q16    = (_Float16*)(ws);                         // 1 MB
    _Float16* k16    = (_Float16*)(ws + (1u << 20));            // 1 MB
    _Float16* v16p   = (_Float16*)(ws + (2u << 20));            // 1 MB
    _Float16* attn16 = (_Float16*)(ws + (3u << 20));            // 1 MB
    _Float16* x16    = (_Float16*)(ws + (4u << 20));            // 2 MB
    _Float16* wg16   = (_Float16*)(ws + (6u << 20));            // 144 KB
    _Float16* wq16   = (_Float16*)(ws + (6u << 20) + (256u << 10)); // 48 KB
    _Float16* wo16   = (_Float16*)(ws + (6u << 20) + (512u << 10)); // 72 KB

    // one-shot fp32 -> f16 conversion of activations and all conv weights
    const int nx  = NB * CIN * HW / 8;       // 131072 vec8
    const int nwg = 64 * CIN * 9 / 8;        // 9216
    const int nwq = 192 * CIN / 8;           // 3072
    const int nwo = 64 * 64 * 9 / 8;         // 4608
    cvt_f32_to_f16_kernel<<<(nx  + 255) / 256, 256, 0, stream>>>(x,         x16,  nx);
    cvt_f32_to_f16_kernel<<<(nwg + 255) / 256, 256, 0, stream>>>(w_general, wg16, nwg);
    cvt_f32_to_f16_kernel<<<(nwq + 255) / 256, 256, 0, stream>>>(w_qkv,     wq16, nwq);
    cvt_f32_to_f16_kernel<<<(nwo + 255) / 256, 256, 0, stream>>>(w_out,     wo16, nwo);

    // qkv 1x1 conv: 512 M-tiles x 3 groups (q/k/v), one wave per 16x64 strip
    qkv_gemm_kernel<<<512 * 3, 32, 0, stream>>>(x16, wq16, b_qkv, q16, k16, v16p);
    // fused attention: 64 (b,head) x 8 query blocks, 8 waves per block
    attn_kernel<<<512, 256, 0, stream>>>(q16, k16, v16p, rel_h, rel_w, attn16);
    // general 3x3 conv -> out channels [0,64)
    conv3x3_wmma_kernel<<<512, 32, 0, stream>>>(x16, 128, wg16, b_general, out, 0);
    // attention 3x3 conv -> out channels [64,128)
    conv3x3_wmma_kernel<<<512, 32, 0, stream>>>(attn16, 64, wo16, b_out, out, 64);
}
